// EncoderLayer_71339406786620
// MI455X (gfx1250) — matrix operand order repaired and checked by reading, not compiled, not run on hardware
//
#include <hip/hip_runtime.h>

#define BATCH  2
#define SEQ    2048
#define TOK    4096      // BATCH*SEQ
#define DMODEL 1024
#define NHEAD  16
#define DHD    64
#define DFFN   4096

typedef __attribute__((ext_vector_type(16))) __bf16 v16bf;
typedef __attribute__((ext_vector_type(8)))  float  v8f;

// ---------------------------------------------------------------------------
// WMMA helpers (CDNA5 wave32, D = A(16x32 bf16) * B(32x16 bf16) + C(16x16 f32))
// ---------------------------------------------------------------------------
static __device__ inline v8f wmma_bf16(v16bf a, v16bf b, v8f c) {
  return __builtin_amdgcn_wmma_f32_16x16x32_bf16(false, a, false, b, (short)0, c,
                                                 false, false);
}

// A-matrix fragment (16x32 bf16). Per ISA: lane<16 holds row M=lane with K
// chunks {0..7, 16..23}; lane>=16 holds row M=lane-16 with chunks {8..15, 24..31}.
static __device__ inline v16bf load_afrag(const __bf16* rowp, int hi) {
  union { v16bf v; uint4 u[2]; } fr;
  fr.u[0] = *(const uint4*)(rowp + hi * 8);
  fr.u[1] = *(const uint4*)(rowp + 16 + hi * 8);
  return fr.v;
}

// B-matrix fragment (32x16 bf16). lane<16: col N=lane, K {0..7, 16..23};
// lane>=16: col N=lane-16, K {8..15, 24..31}. rowp points at the lane's column stored K-major.
static __device__ inline v16bf load_bfrag(const __bf16* rowp, int hi) {
  union { v16bf v; uint4 u[2]; } fr;
  fr.u[0] = *(const uint4*)(rowp + hi * 8);
  fr.u[1] = *(const uint4*)(rowp + 16 + hi * 8);
  return fr.v;
}

// CDNA5 async global->LDS copy (ASYNCcnt tracked; 16B per lane).
// lds_addr = low 32 bits of generic shared pointer.
static __device__ inline void async_copy_b128(unsigned lds_addr, const void* g) {
  asm volatile("global_load_async_to_lds_b128 %0, %1, off"
               :
               : "v"(lds_addr), "v"(g)
               : "memory");
}
static __device__ inline void wait_async() {
  asm volatile("s_wait_asynccnt 0" ::: "memory");
}

// ---------------------------------------------------------------------------
// fp32 -> bf16 converter
// ---------------------------------------------------------------------------
__global__ void cvt_f32_bf16(const float* __restrict__ s, __bf16* __restrict__ d,
                             long n) {
  long i = (long)blockIdx.x * blockDim.x + threadIdx.x;
  long st = (long)gridDim.x * blockDim.x;
  for (; i < n; i += st) d[i] = (__bf16)s[i];
}

// ---------------------------------------------------------------------------
// Tiled bf16 WMMA GEMM: C[M,N] = act(A[M,K] @ W[K,N] + bias)
// 256 threads = 8 waves (2x4), block tile 128x128, BK=32.
// Double-buffered LDS; A tile staged with async global->LDS b128 copies,
// B tile staged via registers (transposed to [n][k] for b128 fragment reads).
// ---------------------------------------------------------------------------
template <bool RELU, bool WF32, bool WBF16>
__global__ __launch_bounds__(256)
void gemm_bf16(const __bf16* __restrict__ A, const __bf16* __restrict__ W,
               const float* __restrict__ bias, float* __restrict__ Cf,
               __bf16* __restrict__ Ch, int M, int N, int K) {
  __shared__ __attribute__((aligned(16))) __bf16 As[2][128][40];
  __shared__ __attribute__((aligned(16))) __bf16 Bs[2][128][40];

  const int tid = threadIdx.x;
  const int lane = tid & 31;
  const int w = tid >> 5;
  const int wm = w >> 2, wn = w & 3;       // 2 x 4 wave grid
  const int hi = lane >> 4, ln = lane & 15;
  const int mb = blockIdx.y * 128, nb = blockIdx.x * 128;
  const int nk = K >> 5;

  // A staging: thread owns row r = tid>>1, 16 bf16 at col cc = (tid&1)*16,
  // copied as two b128 chunks (cc, cc+8) per buffer.
  const int ar = tid >> 1;
  const int acc_ = (tid & 1) * 16;
  const __bf16* gA = A + (size_t)(mb + ar) * K + acc_;
  unsigned ldsA[2][2];
#pragma unroll
  for (int bfi = 0; bfi < 2; ++bfi) {
    ldsA[bfi][0] = (unsigned)(uintptr_t)&As[bfi][ar][acc_];
    ldsA[bfi][1] = (unsigned)(uintptr_t)&As[bfi][ar][acc_ + 8];
  }

  v8f acc[4][2];
#pragma unroll
  for (int i = 0; i < 4; ++i)
#pragma unroll
    for (int j = 0; j < 2; ++j)
#pragma unroll
      for (int r = 0; r < 8; ++r) acc[i][j][r] = 0.f;

  uint2 breg[4];
  // ---- prologue: stage tile 0 ----------------------------------------------
  async_copy_b128(ldsA[0][0], gA);
  async_copy_b128(ldsA[0][1], gA + 8);
#pragma unroll
  for (int i = 0; i < 4; ++i) {
    int c = tid * 4 + i;                  // 0..1023
    int kk = c >> 5, nn = (c & 31) * 4;
    breg[i] = *(const uint2*)(W + (size_t)kk * N + nb + nn);
  }
#pragma unroll
  for (int i = 0; i < 4; ++i) {
    int c = tid * 4 + i;
    int kk = c >> 5, nn = (c & 31) * 4;
    union { uint2 u; __bf16 t[4]; } tb;
    tb.u = breg[i];
#pragma unroll
    for (int j2 = 0; j2 < 4; ++j2) Bs[0][nn + j2][kk] = tb.t[j2];
  }
  wait_async();
  __syncthreads();

  // ---- main k loop (ping-pong, one barrier per step) -----------------------
  for (int kt = 0; kt < nk; ++kt) {
    const int p = kt & 1;
    const bool more = (kt + 1) < nk;
    const int k0n = (kt + 1) << 5;

    if (more) {
      async_copy_b128(ldsA[p ^ 1][0], gA + k0n);
      async_copy_b128(ldsA[p ^ 1][1], gA + k0n + 8);
#pragma unroll
      for (int i = 0; i < 4; ++i) {
        int c = tid * 4 + i;
        int kk = c >> 5, nn = (c & 31) * 4;
        breg[i] = *(const uint2*)(W + (size_t)(k0n + kk) * N + nb + nn);
      }
    }

#pragma unroll
    for (int i = 0; i < 4; ++i) {
      v16bf a = load_afrag(&As[p][wm * 64 + i * 16 + ln][0], hi);
#pragma unroll
      for (int j = 0; j < 2; ++j) {
        v16bf bb = load_bfrag(&Bs[p][wn * 32 + j * 16 + ln][0], hi);
        acc[i][j] = wmma_bf16(a, bb, acc[i][j]);
      }
    }

    if (more) {
#pragma unroll
      for (int i = 0; i < 4; ++i) {
        int c = tid * 4 + i;
        int kk = c >> 5, nn = (c & 31) * 4;
        union { uint2 u; __bf16 t[4]; } tb;
        tb.u = breg[i];
#pragma unroll
        for (int j2 = 0; j2 < 4; ++j2) Bs[p ^ 1][nn + j2][kk] = tb.t[j2];
      }
    }
    wait_async();
    __syncthreads();
  }

  // ---- epilogue (branch-free, bias hoisted) --------------------------------
  const float bc0 = bias[nb + wn * 32 + ln];
  const float bc1 = bias[nb + wn * 32 + 16 + ln];
#pragma unroll
  for (int i = 0; i < 4; ++i)
#pragma unroll
    for (int j = 0; j < 2; ++j) {
      const float bc = j ? bc1 : bc0;
#pragma unroll
      for (int r = 0; r < 8; ++r) {
        int row = mb + wm * 64 + i * 16 + r + 8 * hi;
        int col = nb + wn * 32 + j * 16 + ln;
        float v2 = acc[i][j][r] + bc;
        if (RELU) v2 = fmaxf(v2, 0.f);
        size_t idx = (size_t)row * N + col;
        if (WF32) Cf[idx] = v2;
        if (WBF16) Ch[idx] = (__bf16)v2;
      }
    }
}

// ---------------------------------------------------------------------------
// Flash attention with Music-Transformer relative bias.
// Block = 4 waves = 64 query rows for one (b,h). Wave owns 16 rows.
// srel[l,m] = q_l . E[SEQ-1-l+m]  for m<=l, else 0 (skew+mask of reference).
// ---------------------------------------------------------------------------
__global__ __launch_bounds__(128)
void attn_relpos(const __bf16* __restrict__ q, const __bf16* __restrict__ k,
                 const __bf16* __restrict__ v, const __bf16* __restrict__ e,
                 __bf16* __restrict__ ctx) {
  __shared__ __bf16 Vt[DHD][64];     // V tile transposed: Vt[d][m]
  __shared__ float  Ssc[4][16][64];  // logits staging per wave
  __shared__ float  Rsc[4][16][80];  // rel-pos GEMM staging per wave
  __shared__ __bf16 Pl[4][16][64];   // softmax probabilities (bf16) per wave

  const int lane = threadIdx.x & 31;
  const int wid  = threadIdx.x >> 5;
  const int hi = lane >> 4, ln = lane & 15;
  const int h = blockIdx.y, bb = blockIdx.z;
  const int l0 = blockIdx.x * 64 + wid * 16;

  // Q fragments straight from global (rows contiguous in DH)
  const __bf16* qbase = q + ((size_t)(bb * SEQ + l0 + ln)) * DMODEL + h * DHD;
  const v16bf qf0 = load_afrag(qbase, hi);
  const v16bf qf1 = load_afrag(qbase + 32, hi);

  v8f O[4];
  for (int i = 0; i < 4; ++i)
    for (int r = 0; r < 8; ++r) O[i][r] = 0.f;
  float m_state = -3.0e38f, l_state = 0.f;
  const float scale = 0.125f;   // 1/sqrt(64)

  for (int m0 = 0; m0 < SEQ; m0 += 64) {
    // ---- stage V transposed -------------------------------------------------
    {
      int r = threadIdx.x >> 1;
      int d0 = (threadIdx.x & 1) * 32;
      const __bf16* vp = v + ((size_t)(bb * SEQ + m0 + r)) * DMODEL + h * DHD + d0;
#pragma unroll
      for (int j = 0; j < 32; ++j) Vt[d0 + j][r] = vp[j];
    }
    __syncthreads();

    // ---- R = Q @ E_slice^T (16x80), rel rows base = SEQ-1-(l0+15)+m0 --------
    const int basew = (SEQ - 1) - (l0 + 15) + m0;
#pragma unroll
    for (int ns = 0; ns < 5; ++ns) {
      int eidx = basew + ns * 16 + ln;
      eidx = eidx < 0 ? 0 : (eidx > SEQ - 1 ? SEQ - 1 : eidx);  // OOB rows masked later
      const __bf16* erow = e + (size_t)eidx * DHD;
      v8f r;
      for (int t = 0; t < 8; ++t) r[t] = 0.f;
      r = wmma_bf16(qf0, load_bfrag(erow, hi), r);
      r = wmma_bf16(qf1, load_bfrag(erow + 32, hi), r);
#pragma unroll
      for (int t = 0; t < 8; ++t) Rsc[wid][t + 8 * hi][ns * 16 + ln] = r[t];
    }

    // ---- S = Q @ K^T, add skewed bias, write logits -------------------------
#pragma unroll
    for (int ns = 0; ns < 4; ++ns) {
      const __bf16* krow =
          k + ((size_t)(bb * SEQ + m0 + ns * 16 + ln)) * DMODEL + h * DHD;
      v8f s;
      for (int t = 0; t < 8; ++t) s[t] = 0.f;
      s = wmma_bf16(qf0, load_bfrag(krow, hi), s);
      s = wmma_bf16(qf1, load_bfrag(krow + 32, hi), s);
#pragma unroll
      for (int t = 0; t < 8; ++t) {
        int row = t + 8 * hi;
        int col = ns * 16 + ln;
        int lq = l0 + row, mm = m0 + col;
        // e index = basew + (15-row) + col = SEQ-1-lq+mm  (skew), masked m<=l
        float bias2 = (mm <= lq) ? Rsc[wid][row][(15 - row) + col] : 0.f;
        Ssc[wid][row][col] = (s[t] + bias2) * scale;
      }
    }

    // ---- online softmax: lane handles row ln, cols [32*hi, 32*hi+31] --------
    float vals[32];
    float lmax = -3.0e38f;
#pragma unroll
    for (int j = 0; j < 32; ++j) {
      vals[j] = Ssc[wid][ln][hi * 32 + j];
      lmax = fmaxf(lmax, vals[j]);
    }
    lmax = fmaxf(lmax, __shfl_xor(lmax, 16));
    float newm = fmaxf(m_state, lmax);
    float alpha = __expf(m_state - newm);
    float ls = 0.f;
#pragma unroll
    for (int j = 0; j < 32; ++j) {
      float pr = __expf(vals[j] - newm);
      ls += pr;
      Pl[wid][ln][hi * 32 + j] = (__bf16)pr;
    }
    ls += __shfl_xor(ls, 16);
    l_state = l_state * alpha + ls;
    m_state = newm;
    __syncthreads();

    // ---- O = O*alpha + P @ V ------------------------------------------------
#pragma unroll
    for (int dsb = 0; dsb < 4; ++dsb) {
      v8f o = O[dsb];
#pragma unroll
      for (int t = 0; t < 8; ++t) o[t] *= __shfl(alpha, t + 8 * hi);
#pragma unroll
      for (int mblk = 0; mblk < 2; ++mblk) {
        v16bf pa = load_afrag(&Pl[wid][ln][mblk * 32], hi);
        v16bf vb = load_bfrag(&Vt[dsb * 16 + ln][mblk * 32], hi);
        o = wmma_bf16(pa, vb, o);
      }
      O[dsb] = o;
    }
    __syncthreads();
  }

  // ---- finalize: O / l, store bf16 context ----------------------------------
  const float linv = 1.f / l_state;
#pragma unroll
  for (int dsb = 0; dsb < 4; ++dsb)
#pragma unroll
    for (int t = 0; t < 8; ++t) {
      int row = t + 8 * hi;
      float sc = __shfl(linv, t + 8 * hi);
      ctx[((size_t)(bb * SEQ + l0 + row)) * DMODEL + h * DHD + dsb * 16 + ln] =
          (__bf16)(O[dsb][t] * sc);
    }
}

// ---------------------------------------------------------------------------
// out = LayerNorm(xa + xb) * g + beta ; optional bf16 copy
// ---------------------------------------------------------------------------
__global__ __launch_bounds__(256)
void add_layernorm(const float* __restrict__ xa, const float* __restrict__ xb,
                   const float* __restrict__ g, const float* __restrict__ be,
                   float* __restrict__ of, __bf16* __restrict__ oh) {
  __shared__ float s1[256], s2[256];
  const int row = blockIdx.x, tid = threadIdx.x;
  const float* pa = xa + (size_t)row * DMODEL;
  const float* pb = xb + (size_t)row * DMODEL;
  float a = 0.f, b2 = 0.f;
  for (int i = tid; i < DMODEL; i += 256) {
    float t = pa[i] + pb[i];
    a += t;
    b2 += t * t;
  }
  s1[tid] = a;
  s2[tid] = b2;
  __syncthreads();
  for (int st = 128; st > 0; st >>= 1) {
    if (tid < st) { s1[tid] += s1[tid + st]; s2[tid] += s2[tid + st]; }
    __syncthreads();
  }
  const float mu = s1[0] * (1.f / DMODEL);
  const float var = s2[0] * (1.f / DMODEL) - mu * mu;
  const float rs = rsqrtf(var + 1e-6f);
  for (int i = tid; i < DMODEL; i += 256) {
    float t = (pa[i] + pb[i] - mu) * rs * g[i] + be[i];
    of[(size_t)row * DMODEL + i] = t;
    if (oh) oh[(size_t)row * DMODEL + i] = (__bf16)t;
  }
}

// ---------------------------------------------------------------------------
// Host orchestration
// ---------------------------------------------------------------------------
extern "C" void kernel_launch(void* const* d_in, const int* in_sizes, int n_in,
                              void* d_out, int out_size, void* d_ws, size_t ws_size,
                              hipStream_t stream) {
  (void)in_sizes; (void)n_in; (void)out_size; (void)ws_size;
  const float* x   = (const float*)d_in[0];
  const float* Wq  = (const float*)d_in[1];
  const float* bq  = (const float*)d_in[2];
  const float* Wk  = (const float*)d_in[3];
  const float* bk  = (const float*)d_in[4];
  const float* Wv  = (const float*)d_in[5];
  const float* bv  = (const float*)d_in[6];
  const float* Wo  = (const float*)d_in[7];
  const float* bo  = (const float*)d_in[8];
  const float* E   = (const float*)d_in[9];
  const float* W1  = (const float*)d_in[10];
  const float* b1  = (const float*)d_in[11];
  const float* W2  = (const float*)d_in[12];
  const float* b2  = (const float*)d_in[13];
  const float* g1  = (const float*)d_in[14];
  const float* be1 = (const float*)d_in[15];
  const float* g2  = (const float*)d_in[16];
  const float* be2 = (const float*)d_in[17];
  float* out = (float*)d_out;

  char* base = (char*)d_ws;
  size_t off = 0;
  auto alloc = [&](size_t bytes) -> void* {
    void* r = base + off;
    off += bytes;
    off = (off + 255) & ~(size_t)255;
    return r;
  };
  __bf16* x16  = (__bf16*)alloc((size_t)TOK * DMODEL * 2);
  __bf16* wq16 = (__bf16*)alloc((size_t)DMODEL * DMODEL * 2);
  __bf16* wk16 = (__bf16*)alloc((size_t)DMODEL * DMODEL * 2);
  __bf16* wv16 = (__bf16*)alloc((size_t)DMODEL * DMODEL * 2);
  __bf16* wo16 = (__bf16*)alloc((size_t)DMODEL * DMODEL * 2);
  __bf16* w116 = (__bf16*)alloc((size_t)DMODEL * DFFN * 2);
  __bf16* w216 = (__bf16*)alloc((size_t)DFFN * DMODEL * 2);
  __bf16* e16  = (__bf16*)alloc((size_t)SEQ * DHD * 2);
  __bf16* q16  = (__bf16*)alloc((size_t)TOK * DMODEL * 2);
  __bf16* k16  = (__bf16*)alloc((size_t)TOK * DMODEL * 2);
  __bf16* v16  = (__bf16*)alloc((size_t)TOK * DMODEL * 2);
  __bf16* c16  = (__bf16*)alloc((size_t)TOK * DMODEL * 2);
  float*  attnf = (float*)alloc((size_t)TOK * DMODEL * 4);
  float*  out1f = (float*)alloc((size_t)TOK * DMODEL * 4);
  __bf16* out1h = (__bf16*)alloc((size_t)TOK * DMODEL * 2);
  __bf16* h1h   = (__bf16*)alloc((size_t)TOK * DFFN * 2);
  float*  ffnf  = (float*)alloc((size_t)TOK * DMODEL * 4);

  const dim3 cvtg(1024), cvtb(256);
  cvt_f32_bf16<<<cvtg, cvtb, 0, stream>>>(x,  x16,  (long)TOK * DMODEL);
  cvt_f32_bf16<<<cvtg, cvtb, 0, stream>>>(Wq, wq16, (long)DMODEL * DMODEL);
  cvt_f32_bf16<<<cvtg, cvtb, 0, stream>>>(Wk, wk16, (long)DMODEL * DMODEL);
  cvt_f32_bf16<<<cvtg, cvtb, 0, stream>>>(Wv, wv16, (long)DMODEL * DMODEL);
  cvt_f32_bf16<<<cvtg, cvtb, 0, stream>>>(Wo, wo16, (long)DMODEL * DMODEL);
  cvt_f32_bf16<<<cvtg, cvtb, 0, stream>>>(W1, w116, (long)DMODEL * DFFN);
  cvt_f32_bf16<<<cvtg, cvtb, 0, stream>>>(W2, w216, (long)DFFN * DMODEL);
  cvt_f32_bf16<<<cvtg, cvtb, 0, stream>>>(E,  e16,  (long)SEQ * DHD);

  // q/k/v projections: [4096,1024] = [4096,1024] @ [1024,1024]
  gemm_bf16<false, false, true><<<dim3(DMODEL / 128, TOK / 128), 256, 0, stream>>>(
      x16, wq16, bq, nullptr, q16, TOK, DMODEL, DMODEL);
  gemm_bf16<false, false, true><<<dim3(DMODEL / 128, TOK / 128), 256, 0, stream>>>(
      x16, wk16, bk, nullptr, k16, TOK, DMODEL, DMODEL);
  gemm_bf16<false, false, true><<<dim3(DMODEL / 128, TOK / 128), 256, 0, stream>>>(
      x16, wv16, bv, nullptr, v16, TOK, DMODEL, DMODEL);

  // relative-position flash attention
  attn_relpos<<<dim3(SEQ / 64, NHEAD, BATCH), 128, 0, stream>>>(q16, k16, v16,
                                                                e16, c16);

  // output projection
  gemm_bf16<false, true, false><<<dim3(DMODEL / 128, TOK / 128), 256, 0, stream>>>(
      c16, wo16, bo, attnf, nullptr, TOK, DMODEL, DMODEL);

  // out1 = LN(attn_out + x)
  add_layernorm<<<dim3(TOK), 256, 0, stream>>>(attnf, x, g1, be1, out1f, out1h);

  // FFN
  gemm_bf16<true, false, true><<<dim3(DFFN / 128, TOK / 128), 256, 0, stream>>>(
      out1h, w116, b1, nullptr, h1h, TOK, DFFN, DMODEL);
  gemm_bf16<false, true, false><<<dim3(DMODEL / 128, TOK / 128), 256, 0, stream>>>(
      h1h, w216, b2, ffnf, nullptr, TOK, DMODEL, DFFN);

  // out2 = LN(out1 + ffn) -> d_out (f32)
  add_layernorm<<<dim3(TOK), 256, 0, stream>>>(out1f, ffnf, g2, be2, out,
                                               nullptr);
}
